// AttLoc_82721070121578
// MI455X (gfx1250) — compile-verified, matrix operand order checked by reading, not run on hardware
//
#include <hip/hip_runtime.h>
#include <math.h>

// ---------------------------------------------------------------------------
// Location-aware attention (ESPnet AttLoc) for MI455X / gfx1250, wave32.
// Big GEMM (64000x1056 @ 1056x512) on v_wmma_f32_16x16x32_bf16:
//   - conv-feature term folded into K (K = 1024 enc + 32 conv = 1056)
//   - M=32 rows per WG: 8 WMMAs per k-step reuse 4 B-fragments
//   - explicit ping-pong double buffering (no rotation copies, loads one
//     k-step ahead of the WMMAs that consume them)
//   - tanh + g-dot fused in epilogue; [B,T,512] never touches HBM
// ---------------------------------------------------------------------------

typedef __attribute__((ext_vector_type(16))) __bf16 v16bf;
typedef __attribute__((ext_vector_type(8)))  float  v8f;
typedef __attribute__((ext_vector_type(4)))  float  v4f;

constexpr int B_   = 32;
constexpr int T_   = 2000;
constexpr int ED   = 1024;        // ENC_DIM == DEC_DIM
constexpr int AD   = 512;         // ATT_DIM
constexpr int CC   = 32;          // CONV_CH
constexpr int KW   = 101;         // 2*KSIZE+1
constexpr int OD   = 1024;        // ATT_ODIM
constexpr int KCAT = ED + CC;     // 1056 : stacked K  (enc || conv feature)
constexpr int KKT  = KCAT / 32;   // 33 k-tiles of 32
constexpr int MT   = 32;          // rows per WG in e-GEMM (2 m-tiles)
constexpr int ROWS = B_ * T_;     // 64000 flattened rows
constexpr float SCALE_ = 2.0f;
constexpr int TSEG = 16;          // t-segments for deterministic context reduce

__device__ __forceinline__ __bf16 to_bf16(float f) {
    unsigned u = __float_as_uint(f);
    u += 0x7fffu + ((u >> 16) & 1u);           // round-to-nearest-even
    unsigned short s = (unsigned short)(u >> 16);
    return __builtin_bit_cast(__bf16, s);
}

// Fragment K pattern for 16-bit WMMA operands (16x32 A / 32x16 B), ISA 7.12.2:
// lanes 0-15 hold K=0..7,16..23 ; lanes 16-31 hold K=8..15,24..31.
__device__ __forceinline__ int frag_k(int lane, int i) {
    int kb = (lane >> 4) << 3;
    return (i < 8) ? (kb + i) : (16 + kb + (i - 8));
}

// ---------------------------------------------------------------------------
// Pack a (stacked) f32 weight matrix [K0+.., N] row-major into bf16 WMMA
// B-fragments: block (kk,ntile) = 32 lanes x 16 halves, 32B contiguous/lane.
// ---------------------------------------------------------------------------
__global__ void pack_b(const float* __restrict__ W0, int K0,
                       const float* __restrict__ W1, int N,
                       __bf16* __restrict__ out) {
    int ntiles = N >> 4;
    int kk     = blockIdx.x / ntiles;
    int ntile  = blockIdx.x % ntiles;
    int lane   = threadIdx.x;
    int n      = ntile * 16 + (lane & 15);
    __bf16* dst = out + ((size_t)blockIdx.x * 32 + lane) * 16;
#pragma unroll
    for (int i = 0; i < 16; ++i) {
        int k = kk * 32 + frag_k(lane, i);
        float v = (k < K0) ? W0[(size_t)k * N + n]
                           : W1[(size_t)(k - K0) * N + n];
        dst[i] = to_bf16(v);
    }
}

// decb[b][a] = b_dec[a] + b_enc[a] + b_att[a] + dec_h[b,:] @ W_dec[:,a]
__global__ __launch_bounds__(256)
void dec_kernel(const float* __restrict__ dec_h, const float* __restrict__ W_dec,
                const float* __restrict__ b_dec, const float* __restrict__ b_enc,
                const float* __restrict__ b_att, float* __restrict__ decb) {
    int b = blockIdx.x;
    const float* dh = dec_h + (size_t)b * ED;
    for (int a = threadIdx.x; a < AD; a += 256) {
        float s = b_dec[a] + b_enc[a] + b_att[a];
        for (int k = 0; k < ED; ++k)
            s = fmaf(dh[k], W_dec[(size_t)k * AD + a], s);
        decb[(size_t)b * AD + a] = s;
    }
}

// convout[b][t][c] = sum_j att_prev[b, t+j-50] * conv_w[c*101+j]  (SAME padding)
__global__ __launch_bounds__(256)
void conv_kernel(const float* __restrict__ att_prev,
                 const float* __restrict__ conv_w,
                 float* __restrict__ convout) {
    __shared__ float sw[CC * KW];
    __shared__ float sx[8 + KW - 1];          // 108
    int blk = blockIdx.x;
    int b   = blk / (T_ / 8);
    int t0  = (blk % (T_ / 8)) * 8;
    int tid = threadIdx.x;
    for (int i = tid; i < CC * KW; i += 256) sw[i] = conv_w[i];
    for (int i = tid; i < 8 + KW - 1; i += 256) {
        int t = t0 + i - 50;
        sx[i] = (t >= 0 && t < T_) ? att_prev[(size_t)b * T_ + t] : 0.f;
    }
    __syncthreads();
    int c  = tid & 31;
    int dt = tid >> 5;
    float s = 0.f;
    for (int j = 0; j < KW; ++j) s = fmaf(sx[dt + j], sw[c * KW + j], s);
    convout[((size_t)(b * T_ + t0 + dt)) * CC + c] = s;
}

// ---------------------------------------------------------------------------
// Main kernel: one WG (8 waves) per 32-row tile of the flattened [B*T] axis.
// e[row] = g . tanh( [enc|conv](row,:) @ Wcat + decb[row/T] )  (length-masked)
// ---------------------------------------------------------------------------
__global__ __launch_bounds__(256)
void e_kernel(const float* __restrict__ enc, const float* __restrict__ convout,
              const float* __restrict__ decb, const __bf16* __restrict__ wcat,
              const float* __restrict__ gvec, const int* __restrict__ enc_len,
              float* __restrict__ e_out) {
    // A fragments for 2 m-tiles: [kk][mtile][lane][16] bf16  (66 KB)
    __shared__ __align__(32) __bf16 sA[KKT * 2 * 32 * 16];
    __shared__ float sE[MT];

    int gt   = blockIdx.x;                 // 0 .. ROWS/MT-1  (2000)
    int row0 = gt * MT;
    int tid  = threadIdx.x;
    if (tid < MT) sE[tid] = 0.f;

    // ---- Stage A tile into LDS as bf16 fragments (2112 slots of 32 B) ----
    for (int s = tid; s < KKT * 2 * 32; s += 256) {
        int kk    = s >> 6;                 // /64
        int mtile = (s >> 5) & 1;
        int lane  = s & 31;
        int row   = row0 + mtile * 16 + (lane & 15);
        int kb4   = (lane >> 4) << 1;       // (kb=0|8) as v4f index
        v4f l0, l1, h0, h1;
        if (kk < 32) {
            const v4f* src = (const v4f*)(enc + (size_t)row * ED + kk * 32);
            l0 = __builtin_nontemporal_load(src + kb4);
            l1 = __builtin_nontemporal_load(src + kb4 + 1);
            h0 = __builtin_nontemporal_load(src + kb4 + 4);
            h1 = __builtin_nontemporal_load(src + kb4 + 5);
        } else {                            // conv-feature rows (K = 1024..1055)
            const v4f* src = (const v4f*)(convout + (size_t)row * CC);
            l0 = src[kb4]; l1 = src[kb4 + 1];
            h0 = src[kb4 + 4]; h1 = src[kb4 + 5];
        }
        v16bf tmp;
#pragma unroll
        for (int i = 0; i < 4; ++i) {
            tmp[i]      = to_bf16(l0[i]);
            tmp[4 + i]  = to_bf16(l1[i]);
            tmp[8 + i]  = to_bf16(h0[i]);
            tmp[12 + i] = to_bf16(h1[i]);
        }
        *(v16bf*)&sA[(size_t)s * 16] = tmp;
    }
    __syncthreads();

    int wave = tid >> 5;
    int lane = tid & 31;
    int col0 = wave * 64;                  // 4 n-tiles per wave

    v8f acc[2][4];
#pragma unroll
    for (int mt = 0; mt < 2; ++mt)
#pragma unroll
        for (int nt = 0; nt < 4; ++nt)
            acc[mt][nt] = (v8f){0, 0, 0, 0, 0, 0, 0, 0};

    auto loadfrag = [&](int kk, v16bf (&a)[2], v16bf (&bf)[4]) {
#pragma unroll
        for (int nt = 0; nt < 4; ++nt)
            bf[nt] = *(const v16bf*)&wcat[(((size_t)kk * (AD / 16) + (col0 >> 4) + nt) * 32 + lane) * 16];
        a[0] = *(const v16bf*)&sA[(((size_t)kk * 2 + 0) * 32 + lane) * 16];
        a[1] = *(const v16bf*)&sA[(((size_t)kk * 2 + 1) * 32 + lane) * 16];
    };
    auto compute = [&](v16bf (&a)[2], v16bf (&bf)[4]) {
#pragma unroll
        for (int nt = 0; nt < 4; ++nt) {
            acc[0][nt] = __builtin_amdgcn_wmma_f32_16x16x32_bf16(
                false, a[0], false, bf[nt], (short)0, acc[0][nt], false, false);
            acc[1][nt] = __builtin_amdgcn_wmma_f32_16x16x32_bf16(
                false, a[1], false, bf[nt], (short)0, acc[1][nt], false, false);
        }
    };

    // ---- ping-pong pipelined K loop: 16 double-steps (k=0..31) + tail k=32 --
    v16bf a0[2], b0[4], a1[2], b1[4];
    loadfrag(0, a0, b0);
#pragma unroll 1
    for (int it = 0; it < 16; ++it) {
        loadfrag(2 * it + 1, a1, b1);   // in flight under compute of 2*it
        compute(a0, b0);
        loadfrag(2 * it + 2, a0, b0);   // in flight under compute of 2*it+1
        compute(a1, b1);
    }
    compute(a0, b0);                    // k-tile 32 (loaded in last double-step)

    // ---- Epilogue: + decb, tanh, dot with g, reduce to 32 per-row scalars --
    float epart[2][8];
#pragma unroll
    for (int mt = 0; mt < 2; ++mt)
#pragma unroll
        for (int v = 0; v < 8; ++v) epart[mt][v] = 0.f;

#pragma unroll
    for (int nt = 0; nt < 4; ++nt) {
        int acol = col0 + nt * 16 + (lane & 15);
        float g = gvec[acol];
#pragma unroll
        for (int mt = 0; mt < 2; ++mt) {
#pragma unroll
            for (int v = 0; v < 8; ++v) {
                int row = row0 + mt * 16 + v + ((lane >> 4) << 3);
                float d = decb[(size_t)(row / T_) * AD + acol];
                epart[mt][v] = fmaf(tanhf(acc[mt][nt][v] + d), g, epart[mt][v]);
            }
        }
    }
#pragma unroll
    for (int mt = 0; mt < 2; ++mt)
#pragma unroll
        for (int v = 0; v < 8; ++v) {
            float x = epart[mt][v];
            for (int off = 1; off < 16; off <<= 1) x += __shfl_xor(x, off, 32);
            if ((lane & 15) == 0)
                atomicAdd(&sE[mt * 16 + v + ((lane >> 4) << 3)], x);  // ds_add_f32
        }
    __syncthreads();
    if (tid < MT) {
        int row = row0 + tid;
        int b   = row / T_;
        int t   = row - b * T_;
        e_out[row] = (t >= enc_len[b]) ? -1e15f : sE[tid];
    }
}

__global__ __launch_bounds__(256)
void softmax_kernel(const float* __restrict__ e, float* __restrict__ attn) {
    __shared__ float red[256];
    int b = blockIdx.x, tid = threadIdx.x;
    float mx = -3.4e38f;
    for (int t = tid; t < T_; t += 256) mx = fmaxf(mx, SCALE_ * e[(size_t)b * T_ + t]);
    red[tid] = mx; __syncthreads();
    for (int s = 128; s > 0; s >>= 1) { if (tid < s) red[tid] = fmaxf(red[tid], red[tid + s]); __syncthreads(); }
    mx = red[0]; __syncthreads();
    float sum = 0.f;
    for (int t = tid; t < T_; t += 256) sum += __expf(SCALE_ * e[(size_t)b * T_ + t] - mx);
    red[tid] = sum; __syncthreads();
    for (int s = 128; s > 0; s >>= 1) { if (tid < s) red[tid] += red[tid + s]; __syncthreads(); }
    float inv = 1.f / red[0];
    for (int t = tid; t < T_; t += 256)
        attn[(size_t)b * T_ + t] = __expf(SCALE_ * e[(size_t)b * T_ + t] - mx) * inv;
}

// partial[b][seg][d] = sum_{t in seg} attn[b,t] * enc[b,t,d]
// grid: b(32) x seg(16) = 512 blocks; thread handles 4 d's with b128 NT loads.
__global__ __launch_bounds__(256)
void ctx_partial(const float* __restrict__ enc, const float* __restrict__ attn,
                 float* __restrict__ partial) {
    __shared__ float sw[T_ / TSEG];            // 125 attention weights
    int b   = blockIdx.x >> 4;
    int seg = blockIdx.x & 15;
    int t0  = seg * (T_ / TSEG);
    for (int i = threadIdx.x; i < T_ / TSEG; i += 256)
        sw[i] = attn[(size_t)b * T_ + t0 + i];
    __syncthreads();
    int d4 = threadIdx.x * 4;
    v4f acc = (v4f){0, 0, 0, 0};
    for (int i = 0; i < T_ / TSEG; ++i) {
        const v4f* p = (const v4f*)(enc + ((size_t)(b * T_ + t0 + i)) * ED + d4);
        v4f x = __builtin_nontemporal_load(p);   // streaming: keep L2 for weights
        acc += x * sw[i];
    }
    *(v4f*)(partial + ((size_t)(b * TSEG + seg)) * ED + d4) = acc;
}

__global__ __launch_bounds__(256)
void ctx_reduce(const float* __restrict__ partial, float* __restrict__ c_pre) {
    int i = blockIdx.x * 256 + threadIdx.x;    // 0..B*ED-1
    int b = i >> 10, d = i & 1023;
    float s = 0.f;
    for (int seg = 0; seg < TSEG; ++seg)
        s += partial[((size_t)(b * TSEG + seg)) * ED + d];
    c_pre[i] = s;
}

// c = c_pre @ W_o + b_o  via bf16 WMMA. 128 tiles, 8 waves/WG -> 16 WGs.
__global__ __launch_bounds__(256)
void out_kernel(const float* __restrict__ c_pre, const __bf16* __restrict__ wo,
                const float* __restrict__ b_o, float* __restrict__ out) {
    int wave = threadIdx.x >> 5, lane = threadIdx.x & 31;
    int tile  = blockIdx.x * 8 + wave;             // 0..127
    int mtile = tile >> 6;                         // 0..1
    int ntile = tile & 63;                         // 0..63
    int row   = mtile * 16 + (lane & 15);
    int kb    = (lane >> 4) << 3;
    v8f acc = (v8f){0, 0, 0, 0, 0, 0, 0, 0};
    for (int kk = 0; kk < ED / 32; ++kk) {
        const float* src = c_pre + (size_t)row * ED + kk * 32;
        v16bf a;
#pragma unroll
        for (int i = 0; i < 8; ++i) a[i]     = to_bf16(src[kb + i]);
#pragma unroll
        for (int i = 0; i < 8; ++i) a[8 + i] = to_bf16(src[16 + kb + i]);
        v16bf bf = *(const v16bf*)&wo[(((size_t)kk * (OD / 16) + ntile) * 32 + lane) * 16];
        acc = __builtin_amdgcn_wmma_f32_16x16x32_bf16(
            false, a, false, bf, (short)0, acc, false, false);
    }
    int ncol = ntile * 16 + (lane & 15);
#pragma unroll
    for (int v = 0; v < 8; ++v) {
        int r = mtile * 16 + v + ((lane >> 4) << 3);
        out[(size_t)r * OD + ncol] = acc[v] + b_o[ncol];
    }
}

// ---------------------------------------------------------------------------
extern "C" void kernel_launch(void* const* d_in, const int* in_sizes, int n_in,
                              void* d_out, int out_size, void* d_ws, size_t ws_size,
                              hipStream_t stream) {
    const float* enc_pad  = (const float*)d_in[0];
    const int*   enc_len  = (const int*)  d_in[1];
    const float* dec_h    = (const float*)d_in[2];
    const float* att_prev = (const float*)d_in[3];
    const float* W_enc    = (const float*)d_in[4];
    const float* b_enc    = (const float*)d_in[5];
    const float* W_dec    = (const float*)d_in[6];
    const float* b_dec    = (const float*)d_in[7];
    const float* W_att    = (const float*)d_in[8];
    const float* b_att    = (const float*)d_in[9];
    const float* conv_w   = (const float*)d_in[10];
    const float* gvec     = (const float*)d_in[11];
    const float* W_o      = (const float*)d_in[12];
    const float* b_o      = (const float*)d_in[13];

    char* ws = (char*)d_ws;
    size_t off = 0;
    auto carve = [&](size_t bytes) {
        void* p = ws + off;
        off = (off + bytes + 255) & ~(size_t)255;
        return p;
    };
    __bf16* wcat    = (__bf16*)carve((size_t)KCAT * AD * 2);     // ~1.03 MB
    __bf16* wo_p    = (__bf16*)carve((size_t)ED * OD * 2);       //  2 MB
    float*  decb    = (float*) carve((size_t)B_ * AD * 4);
    float*  convout = (float*) carve((size_t)B_ * T_ * CC * 4);  // ~8.2 MB
    float*  e_buf   = (float*) carve((size_t)B_ * T_ * 4);
    float*  partial = (float*) carve((size_t)B_ * TSEG * ED * 4);//  2 MB
    float*  c_pre   = (float*) carve((size_t)B_ * ED * 4);

    float* out_c    = (float*)d_out;           // [32,1024]
    float* out_attn = out_c + B_ * OD;         // [32,2000]

    pack_b<<<KKT * (AD / 16), 32, 0, stream>>>(W_enc, ED, W_att, AD, wcat);
    pack_b<<<(ED / 32) * (OD / 16), 32, 0, stream>>>(W_o, ED, W_o, OD, wo_p);
    dec_kernel<<<B_, 256, 0, stream>>>(dec_h, W_dec, b_dec, b_enc, b_att, decb);
    conv_kernel<<<B_ * (T_ / 8), 256, 0, stream>>>(att_prev, conv_w, convout);
    e_kernel<<<ROWS / MT, 256, 0, stream>>>(enc_pad, convout, decb, wcat,
                                            gvec, enc_len, e_buf);
    softmax_kernel<<<B_, 256, 0, stream>>>(e_buf, out_attn);
    ctx_partial<<<B_ * TSEG, 256, 0, stream>>>(enc_pad, out_attn, partial);
    ctx_reduce<<<(B_ * ED) / 256, 256, 0, stream>>>(partial, c_pre);
    out_kernel<<<(2 * (OD / 16)) / 8, 256, 0, stream>>>(c_pre, wo_p, b_o, out_c);
}
